// AttnLSTM_70531952935086
// MI455X (gfx1250) — matrix operand order repaired and checked by reading, not compiled, not run on hardware
//
#include <hip/hip_runtime.h>
#include <hip/hip_bf16.h>

// AttnLSTM on gfx1250: bf16 WMMA GEMMs, TDM-staged encoder, blocked xg layout,
// hardware tanh/rcp activations.  B=1024, T=512, F=64, H=32, G=4H=128.

#define Bsz 1024
#define Tsz 512
#define Fsz 64
#define Hsz 32
#define Gsz 128

typedef __attribute__((ext_vector_type(16))) __bf16        v16bf;
typedef __attribute__((ext_vector_type(8)))  __bf16        v8bf;
typedef __attribute__((ext_vector_type(8)))  float         v8f;
typedef __attribute__((ext_vector_type(4)))  unsigned int  u32x4;
typedef __attribute__((ext_vector_type(8)))  int           i32x8;
typedef __attribute__((ext_vector_type(4)))  int           i32x4;

#if defined(__has_builtin)
#if __has_builtin(__builtin_amdgcn_tensor_load_to_lds) && __has_builtin(__builtin_amdgcn_s_wait_tensorcnt)
#define USE_TDM 1
#endif
#endif

// v_tanh_f32 is a CDNA5 hardware transcendental; fall back to exp + v_rcp_f32.
#if defined(__has_builtin)
#if __has_builtin(__builtin_amdgcn_tanhf)
#define HAVE_HW_TANH 1
#endif
#endif

__device__ __forceinline__ float tanhf_(float x) {
#if defined(HAVE_HW_TANH)
    return __builtin_amdgcn_tanhf(x);
#else
    return 1.0f - 2.0f * __builtin_amdgcn_rcpf(__expf(2.0f * x) + 1.0f);
#endif
}
__device__ __forceinline__ float sigmoidf_(float x) {
    return __builtin_fmaf(0.5f, tanhf_(0.5f * x), 0.5f);
}

__device__ __forceinline__ unsigned short f2bfu(float f) {
    return __builtin_bit_cast(unsigned short, (__bf16)f);
}

__device__ __forceinline__ v8f zero8() {
    v8f z = {0.f, 0.f, 0.f, 0.f, 0.f, 0.f, 0.f, 0.f};
    return z;
}

// A operand (16x32 bf16) from fp32 memory (LDS staged), row stride ldr floats.
// ISA 7.12.2: lanes 0-15 row M=lane: V0..3 K=0..7, V4..7 K=16..23; lanes 16-31
// same rows, K=8..15 / 24..31.  m=lane&15, kh=lane>>4.
__device__ __forceinline__ v16bf loadA_f32(const float* s, int ldr, int m, int kh, int kbase) {
    const float* p = s + (size_t)m * ldr + kbase + kh * 8;
    v16bf a;
#pragma unroll
    for (int v = 0; v < 4; ++v) { a[2*v] = (__bf16)p[2*v]; a[2*v+1] = (__bf16)p[2*v+1]; }
    const float* q = p + 16;
#pragma unroll
    for (int v = 0; v < 4; ++v) { a[8+2*v] = (__bf16)q[2*v]; a[9+2*v] = (__bf16)q[2*v+1]; }
    return a;
}

// A operand from LDS bf16 bits (row-major [16][ld]).
__device__ __forceinline__ v16bf loadA_bf(const unsigned short* s, int ld, int m, int kh, int kbase) {
    const unsigned short* p = s + m * ld + kbase + kh * 8;
    v16bf a;
#pragma unroll
    for (int v = 0; v < 4; ++v) {
        a[2*v]   = __builtin_bit_cast(__bf16, p[2*v]);
        a[2*v+1] = __builtin_bit_cast(__bf16, p[2*v+1]);
    }
    const unsigned short* q = p + 16;
#pragma unroll
    for (int v = 0; v < 4; ++v) {
        a[8+2*v] = __builtin_bit_cast(__bf16, q[2*v]);
        a[9+2*v] = __builtin_bit_cast(__bf16, q[2*v+1]);
    }
    return a;
}

// B operand from pre-packed bf16 layout: wp[(op*32 + lane)*16 + e], two b128 loads.
__device__ __forceinline__ v16bf loadB_packed(const unsigned short* __restrict__ wp, int op, int lane) {
    const u32x4* p = (const u32x4*)(wp + ((size_t)op * 32 + lane) * 16);
    v8bf lo = __builtin_bit_cast(v8bf, p[0]);
    v8bf hi = __builtin_bit_cast(v8bf, p[1]);
    return __builtin_shufflevector(lo, hi, 0,1,2,3,4,5,6,7,8,9,10,11,12,13,14,15);
}

__device__ __forceinline__ v8f wmma_bf16(v16bf a, v16bf b, v8f c) {
    return __builtin_amdgcn_wmma_f32_16x16x32_bf16(false, a, false, b, (short)0, c, false, false);
}

// ---------------------------------------------------------------------------
// Kernel 0: pack weights into per-lane WMMA B-operand bf16 layout (run once).
// op indexing: op = ct*nks + ks; element e of lane l: col=ct*16+(l&15),
// k = ks*32 + (e>>3)*16 + (l>>4)*8 + (e&7).
// ---------------------------------------------------------------------------
__global__ __launch_bounds__(256) void k_pack(
    const float* __restrict__ Wenc, const float* __restrict__ Wker,
    const float* __restrict__ Wrec, unsigned short* __restrict__ wencP,
    unsigned short* __restrict__ wkerP, unsigned short* __restrict__ wrecP) {
    const int tid = threadIdx.x;
    for (int i = tid; i < 8 * 32 * 16; i += 256) {      // W_enc: 4 ct x 2 ks
        const int e = i & 15, l = (i >> 4) & 31, op = i >> 9;
        const int ct = op >> 1, ks = op & 1;
        const int col = ct * 16 + (l & 15);
        const int k = ks * 32 + (e >> 3) * 16 + (l >> 4) * 8 + (e & 7);
        wencP[i] = f2bfu(Wenc[k * Fsz + col]);
    }
    for (int i = tid; i < 16 * 32 * 16; i += 256) {     // kernel: 8 ct x 2 ks
        const int e = i & 15, l = (i >> 4) & 31, op = i >> 9;
        const int ct = op >> 1, ks = op & 1;
        const int col = ct * 16 + (l & 15);
        const int k = ks * 32 + (e >> 3) * 16 + (l >> 4) * 8 + (e & 7);
        wkerP[i] = f2bfu(Wker[k * Gsz + col]);
    }
    for (int i = tid; i < 8 * 32 * 16; i += 256) {      // recurrent: 8 ct, K=32
        const int e = i & 15, l = (i >> 4) & 31, ct = i >> 9;
        const int col = ct * 16 + (l & 15);
        const int k = (e >> 3) * 16 + (l >> 4) * 8 + (e & 7);
        wrecP[i] = f2bfu(Wrec[k * Gsz + col]);
    }
}

// ---------------------------------------------------------------------------
// Kernel 1: block = (16 batch rows) x (16 timesteps).  x chunk staged to LDS
// by the Tensor Data Mover (2D D#: 16 rows x 1024 elems, stride T*F), then
// enc = tanh(x@W_enc+b_enc), xg = enc@kernel+bias -> blocked bf16 layout
// xg[t][btile][ct][lane][8] (matches WMMA C layout; b128 coalesced).
// ---------------------------------------------------------------------------
__global__ __launch_bounds__(256) void k_enc_gates(
    const float* __restrict__ x, const float* __restrict__ benc,
    const float* __restrict__ bias, const unsigned short* __restrict__ wencP,
    const unsigned short* __restrict__ wkerP, unsigned short* __restrict__ xg) {
    __shared__ float xS[16][16][Fsz];              // 64 KiB f32 staging
    __shared__ unsigned short encS[8][16 * Fsz];   // 16 KiB per-wave enc (bf16)

    const int tid  = threadIdx.x;
    const int w    = tid >> 5;
    const int lane = tid & 31;
    const int m    = lane & 15;
    const int kh   = lane >> 4;
    const int n    = lane & 15;
    const int btile = blockIdx.x >> 5;   // 0..63
    const int tch   = blockIdx.x & 31;   // 0..31
    const int b0 = btile * 16;
    const int t0 = tch * 16;

#if defined(USE_TDM)
    if (tid < 32) {  // wave 0 issues one TDM load covering the 64 KiB tile
        const unsigned long long ga =
            (unsigned long long)(const void*)(x + ((size_t)b0 * Tsz + t0) * Fsz);
        const unsigned int ldsa = (unsigned int)(size_t)(void*)&xS[0][0][0];
        u32x4 g0;
        g0[0] = 1u;                                   // count=1, user descriptor
        g0[1] = ldsa;                                 // lds_addr
        g0[2] = (unsigned int)ga;                     // global_addr[31:0]
        g0[3] = (unsigned int)((ga >> 32) & 0x01FFFFFFu) | (2u << 30);  // addr[56:32] | type=2
        i32x8 g1;
        g1[0] = (int)(2u << 16);                      // data_size=4B, mask=0
        g1[1] = (int)(32768u << 16);                  // tensor_dim0 lo16 (32768)
        g1[2] = (int)((32768u >> 16) | (1024u << 16));// dim0 hi16 | tensor_dim1 lo16 (1024)
        g1[3] = (int)((1024u >> 16) | (1024u << 16)); // dim1 hi16 | tile_dim0 (1024)
        g1[4] = 16;                                   // tile_dim1=16, tile_dim2=0
        g1[5] = 32768;                                // tensor_dim0_stride lo32 (T*F)
        g1[6] = 0;
        g1[7] = 0;
        i32x4 z4 = {0, 0, 0, 0};
        i32x8 z8 = {0, 0, 0, 0, 0, 0, 0, 0};
        __builtin_amdgcn_tensor_load_to_lds(g0, g1, z4, z4, z8, 0);
        __builtin_amdgcn_s_wait_tensorcnt(0);
    }
#else
    for (int i = tid; i < 16 * 16 * 16; i += 256) {   // 4096 x b128 cooperative copy
        const int f4 = i & 15, tl = (i >> 4) & 15, mb = i >> 8;
        const u32x4* src = (const u32x4*)(x + ((size_t)(b0 + mb) * Tsz + t0 + tl) * Fsz);
        ((u32x4*)&xS[mb][tl][0])[f4] = src[f4];
    }
#endif
    __syncthreads();

    // each wave handles timesteps tl = w and w+8
#pragma unroll
    for (int it = 0; it < 2; ++it) {
        const int tl = w + it * 8;
        const int t  = t0 + tl;

        // GEMM1: enc tile (16 b x 64)
        v16bf a0 = loadA_f32(&xS[0][tl][0], 16 * Fsz, m, kh, 0);
        v16bf a1 = loadA_f32(&xS[0][tl][0], 16 * Fsz, m, kh, 32);
        __syncthreads();   // encS reuse fence (uniform across waves)
#pragma unroll
        for (int ct = 0; ct < 4; ++ct) {
            v8f acc = zero8();
            acc = wmma_bf16(a0, loadB_packed(wencP, ct * 2 + 0, lane), acc);
            acc = wmma_bf16(a1, loadB_packed(wencP, ct * 2 + 1, lane), acc);
            const float bv = benc[ct * 16 + n];
#pragma unroll
            for (int v = 0; v < 8; ++v) {
                const int mm = v + 8 * kh;
                encS[w][mm * Fsz + ct * 16 + n] = f2bfu(tanhf_(acc[v] + bv));
            }
        }
        __syncthreads();

        // GEMM2: xg tile (16 b x 128) at fixed t
        v16bf e0 = loadA_bf(encS[w], Fsz, m, kh, 0);
        v16bf e1 = loadA_bf(encS[w], Fsz, m, kh, 32);
#pragma unroll
        for (int ct = 0; ct < 8; ++ct) {
            const float bv = bias[ct * 16 + n];
            v8f acc = {bv, bv, bv, bv, bv, bv, bv, bv};
            acc = wmma_bf16(e0, loadB_packed(wkerP, ct * 2 + 0, lane), acc);
            acc = wmma_bf16(e1, loadB_packed(wkerP, ct * 2 + 1, lane), acc);
            v8bf o;
#pragma unroll
            for (int v = 0; v < 8; ++v) o[v] = (__bf16)acc[v];
            u32x4* xq = (u32x4*)xg + (((size_t)t * 64 + btile) * 8 + ct) * 32 + lane;
            *xq = __builtin_bit_cast(u32x4, o);
        }
    }
}

// ---------------------------------------------------------------------------
// Kernel 2: LSTM scan.  One wave = 16 batch rows for all T steps; per step
// 8x b128 xg loads + 8x v_wmma_f32_16x16x32_bf16 + gate math; h restaged
// C->A layout via per-wave LDS tile.  8 blocks x 8 waves = B/16 tiles.
// ---------------------------------------------------------------------------
__global__ __launch_bounds__(256) void k_lstm(
    const unsigned short* __restrict__ xg, const unsigned short* __restrict__ wrecP,
    float* __restrict__ rnn) {
    __shared__ unsigned short hS[8][16 * Hsz];

    const int tid  = threadIdx.x;
    const int w    = tid >> 5;
    const int lane = tid & 31;
    const int m    = lane & 15;
    const int kh   = lane >> 4;
    const int n    = lane & 15;
    const int btile = blockIdx.x * 8 + w;
    const int b0    = btile * 16;

    v16bf rB[8];
#pragma unroll
    for (int ct = 0; ct < 8; ++ct) rB[ct] = loadB_packed(wrecP, ct, lane);

    v16bf hA;
#pragma unroll
    for (int i = 0; i < 16; ++i) hA[i] = (__bf16)0.0f;
    v8f c0 = zero8(), c1 = zero8();

    for (int t = 0; t < Tsz; ++t) {
        const u32x4* xq = (const u32x4*)xg + (((size_t)t * 64 + btile) * 8) * 32 + lane;
        if (t + 1 < Tsz) {  // prefetch next step's 4 KiB block (global_prefetch_b8)
            __builtin_prefetch(
                (const char*)((const u32x4*)xg + (((size_t)(t + 1) * 64 + btile) * 8) * 32) +
                    lane * 128, 0, 1);
        }
        v8f z[8];
#pragma unroll
        for (int ct = 0; ct < 8; ++ct) {
            const v8bf zb = __builtin_bit_cast(v8bf, xq[ct * 32]);
            v8f zz;
#pragma unroll
            for (int v = 0; v < 8; ++v) zz[v] = (float)zb[v];
            z[ct] = wmma_bf16(hA, rB[ct], zz);
        }

        v8f h0, h1;
#pragma unroll
        for (int v = 0; v < 8; ++v) {
            const float i0 = sigmoidf_(z[0][v]), f0 = sigmoidf_(z[2][v]);
            const float g0 = tanhf_(z[4][v]),    o0 = sigmoidf_(z[6][v]);
            c0[v] = f0 * c0[v] + i0 * g0;
            h0[v] = o0 * tanhf_(c0[v]);
            const float i1 = sigmoidf_(z[1][v]), f1 = sigmoidf_(z[3][v]);
            const float g1 = tanhf_(z[5][v]),    o1 = sigmoidf_(z[7][v]);
            c1[v] = f1 * c1[v] + i1 * g1;
            h1[v] = o1 * tanhf_(c1[v]);
        }

#pragma unroll
        for (int v = 0; v < 8; ++v) {  // h -> rnn_out [b][t][h]
            const int mm = v + 8 * kh;
            const size_t base = ((size_t)(b0 + mm) * Tsz + t) * Hsz;
            rnn[base + n]      = h0[v];
            rnn[base + 16 + n] = h1[v];
        }

        __syncthreads();
#pragma unroll
        for (int v = 0; v < 8; ++v) {  // restage h (C->A layout)
            const int mm = v + 8 * kh;
            hS[w][mm * Hsz + n]      = f2bfu(h0[v]);
            hS[w][mm * Hsz + 16 + n] = f2bfu(h1[v]);
        }
        __syncthreads();
        hA = loadA_bf(hS[w], Hsz, m, kh, 0);
    }
}

// ---------------------------------------------------------------------------
// Kernel 3: attention softmax pool + decode.  One block per batch element.
// ---------------------------------------------------------------------------
__global__ __launch_bounds__(256) void k_attn(
    const float* __restrict__ rnn, const float* __restrict__ attnW,
    const float* __restrict__ attnB, const float* __restrict__ attnU,
    const float* __restrict__ Wdec, const float* __restrict__ bdec,
    float* __restrict__ out) {
    __shared__ float Ws[Hsz * Hsz];
    __shared__ float bS[Hsz], uS[Hsz];
    __shared__ float sc[Tsz];
    __shared__ float red[256];
    __shared__ float pooled[Hsz];

    const int tid = threadIdx.x;
    const int b   = blockIdx.x;
    const float* hb = rnn + (size_t)b * Tsz * Hsz;

    for (int i = tid; i < Hsz * Hsz; i += 256) Ws[i] = attnW[i];
    if (tid < Hsz) { bS[tid] = attnB[tid]; uS[tid] = attnU[tid]; }
    __syncthreads();

    for (int t = tid; t < Tsz; t += 256) {   // s_t = tanh(h_t@W + b) . u
        const float* h = hb + t * Hsz;
        float hv[Hsz];
#pragma unroll
        for (int k = 0; k < Hsz; ++k) hv[k] = h[k];
        float s = 0.f;
#pragma unroll 4
        for (int j = 0; j < Hsz; ++j) {
            float acc = bS[j];
#pragma unroll
            for (int k = 0; k < Hsz; ++k) acc += hv[k] * Ws[k * Hsz + j];
            s += tanhf_(acc) * uS[j];
        }
        sc[t] = s;
    }
    __syncthreads();

    red[tid] = fmaxf(sc[tid], sc[tid + 256]);
    __syncthreads();
    for (int off = 128; off > 0; off >>= 1) {
        if (tid < off) red[tid] = fmaxf(red[tid], red[tid + off]);
        __syncthreads();
    }
    const float mx = red[0];
    __syncthreads();
    const float e0 = __expf(sc[tid] - mx), e1 = __expf(sc[tid + 256] - mx);
    sc[tid] = e0; sc[tid + 256] = e1;
    red[tid] = e0 + e1;
    __syncthreads();
    for (int off = 128; off > 0; off >>= 1) {
        if (tid < off) red[tid] += red[tid + off];
        __syncthreads();
    }
    const float inv = __builtin_amdgcn_rcpf(red[0]);   // v_rcp_f32
    __syncthreads();

    if (tid < Hsz) {   // pooled[h] = sum_t softmax_t * h_t[h]
        float acc = 0.f;
        for (int t = 0; t < Tsz; ++t) acc += sc[t] * hb[t * Hsz + tid];
        pooled[tid] = acc * inv;
    }
    __syncthreads();

    if (tid < Hsz) {
        red[tid] = hb[(size_t)(Tsz - 1) * Hsz + tid] * Wdec[tid] + pooled[tid] * Wdec[Hsz + tid];
    }
    __syncthreads();
    if (tid == 0) {
        float d = bdec[0];
#pragma unroll
        for (int k = 0; k < Hsz; ++k) d += red[k];
        out[b] = sigmoidf_(d);
    }
}

extern "C" void kernel_launch(void* const* d_in, const int* in_sizes, int n_in,
                              void* d_out, int out_size, void* d_ws, size_t ws_size,
                              hipStream_t stream) {
    const float* x     = (const float*)d_in[0];
    const float* Wenc  = (const float*)d_in[1];
    const float* benc  = (const float*)d_in[2];
    const float* Wker  = (const float*)d_in[3];
    const float* Wrec  = (const float*)d_in[4];
    const float* bias  = (const float*)d_in[5];
    const float* attnW = (const float*)d_in[6];
    const float* attnB = (const float*)d_in[7];
    const float* attnU = (const float*)d_in[8];
    const float* Wdec  = (const float*)d_in[9];
    const float* bdec  = (const float*)d_in[10];
    float* out = (float*)d_out;

    const size_t xg_bytes  = (size_t)Tsz * Bsz * Gsz * 2;   // 128 MiB bf16 blocked
    const size_t rnn_bytes = (size_t)Bsz * Tsz * Hsz * 4;   // 64 MiB f32
    unsigned short* xg    = (unsigned short*)d_ws;
    float*          rnn   = (float*)((char*)d_ws + xg_bytes);
    unsigned short* wencP = (unsigned short*)((char*)d_ws + xg_bytes + rnn_bytes);
    unsigned short* wkerP = wencP + 8 * 32 * 16;
    unsigned short* wrecP = wkerP + 16 * 32 * 16;

    k_pack<<<1, 256, 0, stream>>>(Wenc, Wker, Wrec, wencP, wkerP, wrecP);
    // 64 btiles x 32 t-chunks
    k_enc_gates<<<2048, 256, 0, stream>>>(x, benc, bias, wencP, wkerP, xg);
    // 64 batch tiles / 8 waves per block
    k_lstm<<<8, 256, 0, stream>>>(xg, wrecP, rnn);
    k_attn<<<1024, 256, 0, stream>>>(rnn, attnW, attnB, attnU, Wdec, bdec, out);
}
